// AssociationLoss_11759620456888
// MI455X (gfx1250) — compile-verified, matrix operand order checked by reading, not run on hardware
//
#include <hip/hip_runtime.h>
#include <hip/hip_bf16.h>

// ---------------- problem constants ----------------
#define HWN 4096        // H*W = 64*64 feature positions
#define CCH 256         // channels
#define NB  2           // batch
#define ZTHR  (0.028f*0.028f*512.0f*512.0f*2.0f)   // mask_zone threshold
#define NMSTHR (0.004f*0.004f*512.0f*512.0f*2.0f)  // NMS threshold

typedef __attribute__((ext_vector_type(16))) _Float16 v16h;
typedef __attribute__((ext_vector_type(8)))  _Float16 v8h;
typedef __attribute__((ext_vector_type(8)))  float    v8f;
typedef __attribute__((ext_vector_type(4)))  unsigned int v4u;
typedef __attribute__((ext_vector_type(4)))  int v4i;
typedef __attribute__((ext_vector_type(8)))  int v8i;

#if defined(__AMDGCN__) && __has_builtin(__builtin_amdgcn_tensor_load_to_lds) && \
    __has_builtin(__builtin_amdgcn_s_wait_tensorcnt)
#define USE_TDM 1
#else
#define USE_TDM 0
#endif

// ---------------- 1) per-position feature norms ----------------
__global__ void norms_kernel(const float* __restrict__ x1, const float* __restrict__ x2,
                             float* __restrict__ inv1, float* __restrict__ inv2) {
  int gid = blockIdx.x * 256 + threadIdx.x;          // 2 tensors * 2 batches * 4096
  int p = gid & (HWN - 1);
  int n = (gid >> 12) & 1;
  int which = gid >> 13;
  const float* src = which ? x2 : x1;
  const float* base = src + (size_t)n * CCH * HWN + p;
  float ss = 0.0f;
  for (int c = 0; c < CCH; ++c) { float v = base[(size_t)c * HWN]; ss += v * v; }
  float d = fmaxf(sqrtf(ss), 1e-12f);
  (which ? inv2 : inv1)[n * HWN + p] = 1.0f / d;
}

// ---------------- 2) x1: normalize + transpose -> normalized_x (N,HW,C) + f16 A ----------------
__global__ void xform1_kernel(const float* __restrict__ x1, const float* __restrict__ inv1,
                              float* __restrict__ nx, _Float16* __restrict__ x1h) {
  __shared__ float tile[32][33];
  int p0 = blockIdx.x * 32, c0 = blockIdx.y * 32, n = blockIdx.z;
  const float* src = x1 + (size_t)n * CCH * HWN;
  for (int k = 0; k < 4; ++k) {
    int c = c0 + threadIdx.y + k * 8, p = p0 + threadIdx.x;
    tile[threadIdx.y + k * 8][threadIdx.x] = src[(size_t)c * HWN + p] * inv1[n * HWN + p];
  }
  __syncthreads();
  for (int k = 0; k < 4; ++k) {
    int p = p0 + threadIdx.y + k * 8, c = c0 + threadIdx.x;
    float v = tile[threadIdx.x][threadIdx.y + k * 8];
    size_t o = ((size_t)n * HWN + p) * CCH + c;
    nx[o]  = v;
    x1h[o] = (_Float16)v;
  }
}

// ---------------- 3) x2: normalize -> normalized_ref (N,C,HW) + transposed f16 B^T ----------------
__global__ void xform2_kernel(const float* __restrict__ x2, const float* __restrict__ inv2,
                              float* __restrict__ nref, _Float16* __restrict__ x2h) {
  __shared__ float tile[32][33];
  int p0 = blockIdx.x * 32, c0 = blockIdx.y * 32, n = blockIdx.z;
  const float* src = x2 + (size_t)n * CCH * HWN;
  for (int k = 0; k < 4; ++k) {
    int c = c0 + threadIdx.y + k * 8, p = p0 + threadIdx.x;
    float v = src[(size_t)c * HWN + p] * inv2[n * HWN + p];
    nref[(size_t)n * CCH * HWN + (size_t)c * HWN + p] = v;  // same layout, coalesced
    tile[threadIdx.y + k * 8][threadIdx.x] = v;
  }
  __syncthreads();
  for (int k = 0; k < 4; ++k) {
    int p = p0 + threadIdx.y + k * 8, c = c0 + threadIdx.x;
    x2h[((size_t)n * HWN + p) * CCH + c] = (_Float16)tile[threadIdx.x][threadIdx.y + k * 8];
  }
}

// ---------------- 4) WMMA GEMM: sim[n] = A(HWxC) * B(CxHW), B^T rows contiguous ----------------
#if USE_TDM
// Raw 32-bit LDS offset of a static __shared__ object (generic -> AS(3) -> int).
static __device__ __forceinline__ unsigned lds_offset(const void* p) {
  typedef __attribute__((address_space(3))) const char* lds_cptr;
  return (unsigned)(unsigned long long)(lds_cptr)p;
}

// TDM: load a 128-row x 32-half tile of a (HWN x CCH) f16 row-major tensor into LDS.
// D# group0: count=1, lds_addr, 57b global_addr, type=2.
// D# group1: data_size=1(2B), tensor_dim0=CCH, tensor_dim1=HWN,
//            tile_dim0=32, tile_dim1=128, tensor_dim0_stride=CCH.
static __device__ __forceinline__ void tdm_load_tile(unsigned lds_addr, const _Float16* gptr) {
  unsigned long long ga = (unsigned long long)gptr;
  v4u g0;
  g0.x = 1u;                                             // count=1 (valid user D#)
  g0.y = lds_addr;                                       // LDS byte address
  g0.z = (unsigned)(ga & 0xffffffffu);                   // global_addr[31:0]
  g0.w = (unsigned)((ga >> 32) & 0x01ffffffu) | 0x80000000u;  // [56:32] | type=2
  v8i g1;
  g1[0] = 0x10000;                                       // data_size=1 (2 bytes)
  g1[1] = (int)((unsigned)(CCH & 0xffff) << 16);         // tensor_dim0[15:0] @ [63:48]
  g1[2] = (int)(((CCH >> 16) & 0xffff) | ((unsigned)(HWN & 0xffff) << 16)); // dim0 hi | dim1 lo
  g1[3] = (int)(((HWN >> 16) & 0xffff) | (32u << 16));   // dim1 hi | tile_dim0=32
  g1[4] = 128;                                           // tile_dim1=128, tile_dim2=0
  g1[5] = CCH;                                           // tensor_dim0_stride[31:0]
  g1[6] = 0;
  g1[7] = 0;
  v4i z4 = {0, 0, 0, 0};                                 // groups 2/3 unused (2-D tile)
  v8i z8 = {0, 0, 0, 0, 0, 0, 0, 0};
  // clang-23 / therock-10.0 lane: 6-arg form (v4u, v8i, v4i, v4i, v8i, i32 cpol)
  __builtin_amdgcn_tensor_load_to_lds(g0, g1, z4, z4, z8, 0);
}

// Wave tile 64x64 (4x4 WMMA accumulators), 4 waves -> 128x128 per workgroup.
// TDM double-buffers 128x32 A/B panels into LDS; waves consume via ds_load_b128
// using the CDNA5 16-bit A-fragment layout (lane m=lane&15, K-halves 0-7/16-23
// for lanes 0-15, 8-15/24-31 for lanes 16-31).
__global__ void __launch_bounds__(128)
simgemm_kernel(const _Float16* __restrict__ Ah, const _Float16* __restrict__ Bh,
               float* __restrict__ simout) {
  __shared__ _Float16 ldsA[2][128 * 32];
  __shared__ _Float16 ldsB[2][128 * 32];
  const int lane = threadIdx.x & 31;
  const int wave = threadIdx.x >> 5;
  const int rowStart = blockIdx.y * 128;
  const int colStart = blockIdx.x * 128;
  const int n = blockIdx.z;
  const _Float16* A = Ah + (size_t)n * HWN * CCH + (size_t)rowStart * CCH;
  const _Float16* B = Bh + (size_t)n * HWN * CCH + (size_t)colStart * CCH;
  const int m  = lane & 15;
  const int kh = (lane >> 4) << 3;   // 0 or 8
  const int ra = (wave >> 1) * 64;   // wave's rows within the 128-row panel
  const int ca = (wave & 1) * 64;    // wave's cols within the 128-col panel

  v8f acc[4][4] = {};
  if (wave == 0) {                   // TDM ignores EXEC; issue once per workgroup
    tdm_load_tile(lds_offset(&ldsA[0][0]), A);
    tdm_load_tile(lds_offset(&ldsB[0][0]), B);
  }
  int cur = 0;
  for (int k0 = 0; k0 < CCH; k0 += 32, cur ^= 1) {
    if (wave == 0) {
      if (k0 + 32 < CCH) {
        tdm_load_tile(lds_offset(&ldsA[cur ^ 1][0]), A + (k0 + 32));
        tdm_load_tile(lds_offset(&ldsB[cur ^ 1][0]), B + (k0 + 32));
        __builtin_amdgcn_s_wait_tensorcnt(2);  // the 2 just-issued may stay in flight
      } else {
        __builtin_amdgcn_s_wait_tensorcnt(0);
      }
    }
    __syncthreads();                 // publish TDM-written LDS to all waves
    const _Float16* As = &ldsA[cur][0];
    const _Float16* Bs = &ldsB[cur][0];
    v16h a[4], b[4];
#pragma unroll
    for (int i = 0; i < 4; ++i) {
      const _Float16* pa = As + (ra + i * 16 + m) * 32 + kh;
      union { v16h w; v8h h[2]; } ua;
      ua.h[0] = *(const v8h*)pa;
      ua.h[1] = *(const v8h*)(pa + 16);
      a[i] = ua.w;
      const _Float16* pb = Bs + (ca + i * 16 + m) * 32 + kh;
      union { v16h w; v8h h[2]; } ub;
      ub.h[0] = *(const v8h*)pb;
      ub.h[1] = *(const v8h*)(pb + 16);
      b[i] = ub.w;
    }
#pragma unroll
    for (int i = 0; i < 4; ++i)
#pragma unroll
      for (int j = 0; j < 4; ++j)
        acc[i][j] = __builtin_amdgcn_wmma_f32_16x16x32_f16(
            false, a[i], false, b[j], (short)0, acc[i][j], false, false);
    __syncthreads();                 // all reads done before buffer is refilled
  }

  float* outp = simout + (size_t)n * HWN * HWN;
  const int rhalf = (lane >> 4) << 3;   // M offset 0 or 8 per C/D layout
  const int cn = lane & 15;
#pragma unroll
  for (int i = 0; i < 4; ++i)
#pragma unroll
    for (int j = 0; j < 4; ++j) {
      float* po = outp + (size_t)(rowStart + ra + i * 16 + rhalf) * HWN +
                  colStart + ca + j * 16 + cn;
#pragma unroll
      for (int r = 0; r < 8; ++r) po[(size_t)r * HWN] = acc[i][j][r];
    }
}
#else  // fallback: direct global loads (previous working version)
__global__ void __launch_bounds__(128)
simgemm_kernel(const _Float16* __restrict__ Ah, const _Float16* __restrict__ Bh,
               float* __restrict__ simout) {
  const int lane = threadIdx.x & 31;
  const int wave = threadIdx.x >> 5;
  const int rowBase = blockIdx.y * 128 + (wave >> 1) * 64;
  const int colBase = blockIdx.x * 128 + (wave & 1) * 64;
  const int n = blockIdx.z;
  const _Float16* A = Ah + (size_t)n * HWN * CCH;
  const _Float16* B = Bh + (size_t)n * HWN * CCH;
  const int m  = lane & 15;
  const int kh = (lane >> 4) << 3;

  v8f acc[4][4] = {};
  for (int k0 = 0; k0 < CCH; k0 += 32) {
    v16h a[4], b[4];
#pragma unroll
    for (int i = 0; i < 4; ++i) {
      const _Float16* pa = A + (size_t)(rowBase + i * 16 + m) * CCH + (k0 + kh);
      union { v16h w; v8h h[2]; } ua;
      ua.h[0] = *(const v8h*)pa;
      ua.h[1] = *(const v8h*)(pa + 16);
      a[i] = ua.w;
      const _Float16* pb = B + (size_t)(colBase + i * 16 + m) * CCH + (k0 + kh);
      union { v16h w; v8h h[2]; } ub;
      ub.h[0] = *(const v8h*)pb;
      ub.h[1] = *(const v8h*)(pb + 16);
      b[i] = ub.w;
    }
#pragma unroll
    for (int i = 0; i < 4; ++i)
#pragma unroll
      for (int j = 0; j < 4; ++j)
        acc[i][j] = __builtin_amdgcn_wmma_f32_16x16x32_f16(
            false, a[i], false, b[j], (short)0, acc[i][j], false, false);
  }

  float* outp = simout + (size_t)n * HWN * HWN;
  const int rhalf = (lane >> 4) << 3;
  const int cn = lane & 15;
#pragma unroll
  for (int i = 0; i < 4; ++i)
#pragma unroll
    for (int j = 0; j < 4; ++j) {
      float* po = outp + (size_t)(rowBase + i * 16 + rhalf) * HWN + colBase + j * 16 + cn;
#pragma unroll
      for (int r = 0; r < 8; ++r) po[(size_t)r * HWN] = acc[i][j][r];
    }
}
#endif

// ---------------- 5) batch mean + transpose -> sim_mat_12, sim_mat_21 ----------------
__global__ void mean_tr_kernel(const float* __restrict__ simfull,
                               float* __restrict__ sim12, float* __restrict__ sim21) {
  __shared__ float t[32][33];
  int q0 = blockIdx.x * 32, p0 = blockIdx.y * 32;
  for (int k = 0; k < 4; ++k) {
    int p = p0 + threadIdx.y + k * 8, q = q0 + threadIdx.x;
    float v = 0.5f * (simfull[(size_t)p * HWN + q] +
                      simfull[(size_t)HWN * HWN + (size_t)p * HWN + q]);
    sim12[(size_t)p * HWN + q] = v;
    t[threadIdx.y + k * 8][threadIdx.x] = v;
  }
  __syncthreads();
  for (int k = 0; k < 4; ++k) {
    int q = q0 + threadIdx.y + k * 8, p = p0 + threadIdx.x;
    sim21[(size_t)q * HWN + p] = t[threadIdx.x][threadIdx.y + k * 8];
  }
}

// ---------------- 6) keypoint distance matrix + zone mask ----------------
__global__ void pdist_kernel(const float* __restrict__ fkp1, const float* __restrict__ fkp2,
                             float* __restrict__ dmat, float* __restrict__ maskz) {
  int gid = blockIdx.x;                 // 16 * 4096 blocks
  int i = gid >> 4;
  int j = (gid & 15) * 256 + threadIdx.x;
  float ax = fkp1[2 * i], ay = fkp1[2 * i + 1];
  float bx = fkp2[2 * j], by = fkp2[2 * j + 1];
  float dv = fabsf(ax * ax + ay * ay + bx * bx + by * by - 2.0f * (ax * bx + ay * by));
  dmat[(size_t)i * HWN + j]  = dv;
  maskz[(size_t)i * HWN + j] = (dv < ZTHR) ? 1.0f : 0.0f;
}

// ---------------- 7) per-row association: top-2, NMS, re-top-2, masks ----------------
struct Top2 { float v0; int i0; float v1; int i1; };

static __device__ __forceinline__ void t2_insert(Top2& t, float v, int i) {
  if (v > t.v0 || (v == t.v0 && i < t.i0)) {
    t.v1 = t.v0; t.i1 = t.i0; t.v0 = v; t.i0 = i;
  } else if (v > t.v1 || (v == t.v1 && i < t.i1)) {
    t.v1 = v; t.i1 = i;
  }
}

// dir12: simdir=sim12, pR=fkp1, pJ=fkp2 ; dir21: simdir=sim21, pR=fkp2, pJ=fkp1
__global__ void associate_kernel(const float* __restrict__ simdir,
                                 const float* __restrict__ pR, const float* __restrict__ pJ,
                                 int* __restrict__ idxOut, int* __restrict__ maskOutI,
                                 float* __restrict__ simv0Out, float* __restrict__ maskOutF) {
  __shared__ float srow[HWN];
  __shared__ Top2 red[256];
  const int r = blockIdx.x, tid = threadIdx.x;
  const float rx = pR[2 * r], ry = pR[2 * r + 1];
  const float rr = rx * rx + ry * ry;
  for (int j = tid; j < HWN; j += 256) {
    float bx = pJ[2 * j], by = pJ[2 * j + 1];
    float dz = fabsf(rr + bx * bx + by * by - 2.0f * (rx * bx + ry * by));
    srow[j] = simdir[(size_t)r * HWN + j] * ((dz < ZTHR) ? 1.0f : 0.0f);
  }
  __syncthreads();

  Top2 t; t.v0 = -3.4e38f; t.v1 = -3.4e38f; t.i0 = 0x7fffffff; t.i1 = 0x7fffffff;
  for (int j = tid; j < HWN; j += 256) t2_insert(t, srow[j], j);
  red[tid] = t;
  __syncthreads();
  for (int s = 128; s; s >>= 1) {
    if (tid < s) { Top2 o = red[tid + s]; t2_insert(red[tid], o.v0, o.i0); t2_insert(red[tid], o.v1, o.i1); }
    __syncthreads();
  }
  const int idx0 = red[0].i0;         // pre-NMS argmax (mid/max index)
  __syncthreads();

  const float fx = pJ[2 * idx0], fy = pJ[2 * idx0 + 1];
  const float ff = fx * fx + fy * fy;
  t.v0 = -3.4e38f; t.v1 = -3.4e38f; t.i0 = 0x7fffffff; t.i1 = 0x7fffffff;
  for (int j = tid; j < HWN; j += 256) {
    float bx = pJ[2 * j], by = pJ[2 * j + 1];
    float dt = fabsf(ff + bx * bx + by * by - 2.0f * (fx * bx + fy * by));
    float nm = ((dt >= NMSTHR) || (dt == 0.0f)) ? 1.0f : 0.0f;
    t2_insert(t, srow[j] * nm, j);
  }
  red[tid] = t;
  __syncthreads();
  for (int s = 128; s; s >>= 1) {
    if (tid < s) { Top2 o = red[tid + s]; t2_insert(red[tid], o.v0, o.i0); t2_insert(red[tid], o.v1, o.i1); }
    __syncthreads();
  }
  if (tid == 0) {
    float v0 = red[0].v0, v1 = red[0].v1;
    int msk = (v1 < v0 * 0.995f) && (v0 > 0.75f);
    idxOut[r] = idx0;
    maskOutI[r] = msk;
    simv0Out[r] = v0;
    if (maskOutF) maskOutF[r] = (float)msk;
  }
}

// ---------------- 8) data-dependent head outputs (device-computed offsets) ----------------
__global__ void finalize_kernel(const int* __restrict__ mid_idx, const int* __restrict__ max_idx,
                                const int* __restrict__ mask12, const int* __restrict__ mask21,
                                int* __restrict__ outi) {
  __shared__ int sidx[HWN];
  __shared__ int sidx2[HWN];
  __shared__ int mflag[HWN];
  __shared__ int counts[257];
  const int tid = threadIdx.x;
  const int base = tid * 16;

  // compact mask12 positions -> sidx (== jax top_k of mask12 as float)
  int c = 0;
  for (int i = 0; i < 16; ++i) c += (mask12[base + i] != 0);
  counts[tid] = c;
  __syncthreads();
  if (tid == 0) { int acc = 0; for (int t = 0; t < 256; ++t) { int cc = counts[t]; counts[t] = acc; acc += cc; } counts[256] = acc; }
  __syncthreads();
  const int k12 = counts[256];
  { int off = counts[tid]; for (int i = 0; i < 16; ++i) if (mask12[base + i]) sidx[off++] = base + i; }
  __syncthreads();

  // compact mask21 positions -> sidx2
  c = 0;
  for (int i = 0; i < 16; ++i) c += (mask21[base + i] != 0);
  counts[tid] = c;
  __syncthreads();
  if (tid == 0) { int acc = 0; for (int t = 0; t < 256; ++t) { int cc = counts[t]; counts[t] = acc; acc += cc; } counts[256] = acc; }
  __syncthreads();
  const int k21 = counts[256];
  { int off = counts[tid]; for (int i = 0; i < 16; ++i) if (mask21[base + i]) sidx2[off++] = base + i; }
  __syncthreads();

  // output 1: indices = signed21[mid_indices_valid] ; m flag
  for (int i = tid; i < k12; i += 256) {
    int miv = mid_idx[sidx[i]];
    int s21 = (max_idx[miv] + 1) * (mask21[miv] ? 1 : -1) - 1;
    outi[i] = s21;
    mflag[i] = (sidx[i] == s21) ? 1 : 0;
  }
  // outputs 2,3: max_indices, mid_indices
  for (int r = tid; r < HWN; r += 256) {
    outi[k12 + r] = max_idx[r];
    outi[k12 + HWN + r] = mid_idx[r];
  }
  __syncthreads();

  // output 4: index_valid = sidx[i] where m
  int chunk = (k12 + 255) >> 8;
  int b0 = tid * chunk, b1 = b0 + chunk;
  if (b0 > k12) b0 = k12;
  if (b1 > k12) b1 = k12;
  c = 0;
  for (int i = b0; i < b1; ++i) c += (mflag[i] != 0);
  counts[tid] = c;
  __syncthreads();
  if (tid == 0) { int acc = 0; for (int t = 0; t < 256; ++t) { int cc = counts[t]; counts[t] = acc; acc += cc; } counts[256] = acc; }
  __syncthreads();
  const int kv = counts[256];
  { int off = counts[tid]; for (int i = b0; i < b1; ++i) if (mflag[i]) outi[k12 + 2 * HWN + off++] = sidx[i]; }
  __syncthreads();

  // m2 and output 5: index_valid2 = sidx2[i] where sidx2[i]==signed12[max_indices_valid[i]]
  for (int i = tid; i < k21; i += 256) {
    int mav = max_idx[sidx2[i]];
    int s12 = (mid_idx[mav] + 1) * (mask12[mav] ? 1 : -1) - 1;
    mflag[i] = (sidx2[i] == s12) ? 1 : 0;
  }
  __syncthreads();
  chunk = (k21 + 255) >> 8;
  b0 = tid * chunk; b1 = b0 + chunk;
  if (b0 > k21) b0 = k21;
  if (b1 > k21) b1 = k21;
  c = 0;
  for (int i = b0; i < b1; ++i) c += (mflag[i] != 0);
  counts[tid] = c;
  __syncthreads();
  if (tid == 0) { int acc = 0; for (int t = 0; t < 256; ++t) { int cc = counts[t]; counts[t] = acc; acc += cc; } counts[256] = acc; }
  __syncthreads();
  { int off = counts[tid]; for (int i = b0; i < b1; ++i) if (mflag[i]) outi[k12 + 2 * HWN + kv + off++] = sidx2[i]; }
}

// ---------------- host launcher ----------------
extern "C" void kernel_launch(void* const* d_in, const int* in_sizes, int n_in,
                              void* d_out, int out_size, void* d_ws, size_t ws_size,
                              hipStream_t stream) {
  (void)in_sizes; (void)n_in; (void)ws_size;
  const float* x1   = (const float*)d_in[0];
  const float* x2   = (const float*)d_in[1];
  const float* fkp1 = (const float*)d_in[2];
  const float* fkp2 = (const float*)d_in[3];
  float* out = (float*)d_out;

  // Fixed-size tail outputs (return order 6..15); head outputs are data-dependent
  // sizes and written with device-computed offsets by finalize_kernel.
  const long long HWL = HWN;
  const long long T = 3 * HWL * HWL                 // sim_mat_12, sim_mat_21, mask_zone
                    + 2 * (long long)NB * HWL * CCH // normalized_x, normalized_ref
                    + HWL * HWL                     // d
                    + 3 * HWL                       // mask12, associated_sim, max_sim
                    + (long long)NB * HWL * HWL;    // sim
  long long tailBase = (long long)out_size - T;
  if (tailBase < 0) tailBase = 0;

  float* sim12     = out + tailBase;
  float* sim21     = sim12 + (size_t)HWL * HWL;
  float* maskz     = sim21 + (size_t)HWL * HWL;
  float* nx        = maskz + (size_t)HWL * HWL;
  float* nref      = nx + (size_t)NB * HWL * CCH;
  float* dmat      = nref + (size_t)NB * HWL * CCH;
  float* mask12o   = dmat + (size_t)HWL * HWL;
  float* assoc_sim = mask12o + HWL;
  float* max_sim   = assoc_sim + HWL;
  float* simfull   = max_sim + HWL;

  // workspace: f16 operands + norms + association scratch (~8.3 MB)
  char* ws = (char*)d_ws;
  _Float16* x1h = (_Float16*)ws;
  _Float16* x2h = x1h + (size_t)NB * HWL * CCH;
  float* inv1 = (float*)(x2h + (size_t)NB * HWL * CCH);
  float* inv2 = inv1 + NB * HWN;
  int* mid_idx = (int*)(inv2 + NB * HWN);
  int* max_idx = mid_idx + HWN;
  int* mask12i = max_idx + HWN;
  int* mask21i = mask12i + HWN;

  norms_kernel<<<dim3(64), dim3(256), 0, stream>>>(x1, x2, inv1, inv2);
  xform1_kernel<<<dim3(HWN / 32, CCH / 32, NB), dim3(32, 8), 0, stream>>>(x1, inv1, nx, x1h);
  xform2_kernel<<<dim3(HWN / 32, CCH / 32, NB), dim3(32, 8), 0, stream>>>(x2, inv2, nref, x2h);
  simgemm_kernel<<<dim3(HWN / 128, HWN / 128, NB), dim3(128), 0, stream>>>(x1h, x2h, simfull);
  mean_tr_kernel<<<dim3(HWN / 32, HWN / 32), dim3(32, 8), 0, stream>>>(simfull, sim12, sim21);
  pdist_kernel<<<dim3((HWN / 256) * HWN), dim3(256), 0, stream>>>(fkp1, fkp2, dmat, maskz);
  associate_kernel<<<dim3(HWN), dim3(256), 0, stream>>>(sim12, fkp1, fkp2,
                                                        mid_idx, mask12i, assoc_sim, mask12o);
  associate_kernel<<<dim3(HWN), dim3(256), 0, stream>>>(sim21, fkp2, fkp1,
                                                        max_idx, mask21i, max_sim, nullptr);
  finalize_kernel<<<dim3(1), dim3(256), 0, stream>>>(mid_idx, max_idx, mask12i, mask21i,
                                                     (int*)d_out);
}
